// Attn_8804682957098
// MI455X (gfx1250) — compile-verified, matrix operand order checked by reading, not run on hardware
//
#include <hip/hip_runtime.h>
#include <math.h>

typedef __attribute__((ext_vector_type(16))) __bf16 v16bf;
typedef __attribute__((ext_vector_type(8)))  __bf16 v8bf;
typedef __attribute__((ext_vector_type(8)))  float  v8f;

#define WAVES 4

// ---------------- WMMA fragment loader (wave32, 16x16x32 bf16) -----------------
// A/B-frag from a row-major [R][C] bf16 source where C is the K dimension.
// lane l: row = r0 + (l&15); K elements = koff+0..7 and koff+16..23, koff=(l>>4)*8.
__device__ __forceinline__ v16bf ld_frag_bf16(const __bf16* __restrict__ src, int ld,
                                              int r0, int c0) {
  const int lane = threadIdx.x & 31;
  const __bf16* p = src + (size_t)(r0 + (lane & 15)) * ld + c0 + ((lane >> 4) << 3);
  v16bf f;
#pragma unroll
  for (int i = 0; i < 8; ++i) f[i] = p[i];
#pragma unroll
  for (int i = 0; i < 8; ++i) f[8 + i] = p[16 + i];
  return f;
}

__device__ __forceinline__ v8f wmma_bf16(v16bf a, v16bf b, v8f c) {
  return __builtin_amdgcn_wmma_f32_16x16x32_bf16(false, a, false, b, (short)0, c,
                                                 false, false);
}

// ---------------- Kernel 0: one-shot f32 -> bf16 conversion --------------------
// 8 elements/thread: 2x b128 loads, 4x v_cvt_pk_bf16_f32, 1x b128 store.
__global__ __launch_bounds__(256) void cvt_bf16_kernel(const float* __restrict__ src,
                                                       __bf16* __restrict__ dst,
                                                       int n8) {
  int i = blockIdx.x * blockDim.x + threadIdx.x;
  if (i >= n8) return;
  const float4* s = (const float4*)src + (size_t)i * 2;
  float4 a = s[0], b = s[1];
  v8bf d;
  d[0] = (__bf16)a.x; d[1] = (__bf16)a.y; d[2] = (__bf16)a.z; d[3] = (__bf16)a.w;
  d[4] = (__bf16)b.x; d[5] = (__bf16)b.y; d[6] = (__bf16)b.z; d[7] = (__bf16)b.w;
  *((v8bf*)dst + i) = d;
}

// ---------------- Kernel 1: QKV projection + RoPE ------------------------------
// qkv = x(4096x1024) @ w_qkv^T ; 16x64 output tile per wave (A-frag reused 4x).
// q,k -> [b][h][n][d] ; v -> transposed [b][h][d][n] for contiguous P@V B-frags.
__global__ __launch_bounds__(128) void qkv_rope_kernel(
    const __bf16* __restrict__ xb, const float* __restrict__ freqs,
    const __bf16* __restrict__ wb,
    __bf16* __restrict__ qb, __bf16* __restrict__ kb, __bf16* __restrict__ vbT) {
  const int lane = threadIdx.x & 31;
  const int wave = threadIdx.x >> 5;
  const int tile = blockIdx.x * WAVES + wave;        // 0 .. 12287
  const int NT = 48;                                 // 3072/64
  const int mt = tile / NT, nt = tile % NT;
  const int row0 = mt * 16, col0 = nt * 64;

  v8f acc[4] = {};
  for (int k0 = 0; k0 < 1024; k0 += 32) {
    v16bf a = ld_frag_bf16(xb, 1024, row0, k0);
#pragma unroll
    for (int j = 0; j < 4; ++j) {
      v16bf b = ld_frag_bf16(wb, 1024, col0 + j * 16, k0);  // B[k][n]=w[col][k]
      acc[j] = wmma_bf16(a, b, acc[j]);
    }
  }

  const int part = col0 >> 10;                       // 0=q 1=k 2=v (uniform)
  const int head = (col0 & 1023) >> 6;               // uniform per tile
  const int col  = lane & 15;
  const int grp  = (lane >> 4) << 3;

#pragma unroll
  for (int j = 0; j < 4; ++j) {
    const int d = j * 16 + col;
#pragma unroll
    for (int r = 0; r < 8; ++r) {
      const int row  = row0 + r + grp;               // 0..4095
      const int bidx = row >> 11;
      const int npos = row & 2047;
      float val = acc[j][r];
      if (part < 2) {                                // RoPE on q,k
        float partner = __shfl_xor(val, 1, 32);      // paired d lives in lane^1
        float f  = freqs[npos * 32 + (d >> 1)];
        float cv = cosf(f), sv = sinf(f);
        val = (d & 1) ? (partner * sv + val * cv) : (val * cv - partner * sv);
        __bf16* dst = (part == 0) ? qb : kb;
        dst[((size_t)((bidx * 16 + head) * 2048 + npos)) * 64 + d] = (__bf16)val;
      } else {                                       // V stored transposed [d][n]
        vbT[((size_t)((bidx * 16 + head) * 64 + d)) * 2048 + npos] = (__bf16)val;
      }
    }
  }
}

// ---------------- Kernel 2: flash attention, one 16-row q tile per wave --------
__global__ __launch_bounds__(128) void flash_attn_kernel(
    const __bf16* __restrict__ qb, const __bf16* __restrict__ kb,
    const __bf16* __restrict__ vbT, __bf16* __restrict__ ob) {
  __shared__ __bf16 pstage[WAVES][16 * 32];
  const int lane = threadIdx.x & 31;
  const int wave = threadIdx.x >> 5;
  const int gid  = blockIdx.x * WAVES + wave;        // 0..4095
  const int qt = gid & 127;
  const int h  = (gid >> 7) & 15;
  const int b  = gid >> 11;
  const size_t base = (size_t)(b * 16 + h) * 2048 * 64;
  const __bf16* Q  = qb  + base;
  const __bf16* K  = kb  + base;
  const __bf16* Vt = vbT + base;                     // [d][n], ld = 2048
  const int row0 = qt * 16;
  const int grp  = (lane >> 4) << 3;
  const int col  = lane & 15;

  v16bf qa0 = ld_frag_bf16(Q, 64, row0, 0);
  v16bf qa1 = ld_frag_bf16(Q, 64, row0, 32);

  float m[8], l[8];
#pragma unroll
  for (int r = 0; r < 8; ++r) { m[r] = -1e30f; l[r] = 0.0f; }
  v8f o0 = {}, o1 = {}, o2 = {}, o3 = {};
  __bf16* ps = &pstage[wave][0];

  for (int kt = 0; kt < 2048; kt += 32) {
    // prefetch next K tile and next Vt columns into cache (global_prefetch_b8)
    __builtin_prefetch(&K[(size_t)(kt + 32) * 64], 0, 1);
    __builtin_prefetch(&Vt[(size_t)kt + 32], 0, 1);

    // S = Q K^T * scale ; two 16x16 subtiles (cols kt..kt+15, kt+16..kt+31)
    v8f s0 = {}, s1 = {};
    s0 = wmma_bf16(qa0, ld_frag_bf16(K, 64, kt,      0),  s0);
    s0 = wmma_bf16(qa1, ld_frag_bf16(K, 64, kt,      32), s0);
    s1 = wmma_bf16(qa0, ld_frag_bf16(K, 64, kt + 16, 0),  s1);
    s1 = wmma_bf16(qa1, ld_frag_bf16(K, 64, kt + 16, 32), s1);

#pragma unroll
    for (int r = 0; r < 8; ++r) {
      float a0 = s0[r] * 0.125f;                     // SCALE = 64^-0.5
      float a1 = s1[r] * 0.125f;
      float mx = fmaxf(a0, a1);
#pragma unroll
      for (int off = 1; off < 16; off <<= 1)
        mx = fmaxf(mx, __shfl_xor(mx, off, 32));     // row-max across half-group
      float mnew = fmaxf(m[r], mx);
      float corr = __expf(m[r] - mnew);
      float p0 = __expf(a0 - mnew);
      float p1 = __expf(a1 - mnew);
      float rs = p0 + p1;
#pragma unroll
      for (int off = 1; off < 16; off <<= 1)
        rs += __shfl_xor(rs, off, 32);               // row-sum
      l[r] = l[r] * corr + rs;
      m[r] = mnew;
      o0[r] *= corr; o1[r] *= corr; o2[r] *= corr; o3[r] *= corr;
      // stage P row-major [16][32] in LDS (C-layout -> A-layout relayout)
      ps[(r + grp) * 32 + col]      = (__bf16)p0;
      ps[(r + grp) * 32 + 16 + col] = (__bf16)p1;
    }
    asm volatile("s_wait_dscnt 0x0" ::: "memory");   // LDS RAW, same wave
    v16bf pa = ld_frag_bf16(ps, 32, 0, 0);           // P as 16x32 A-frag
    // O += P @ V : B[kk][n] = V[kt+kk][d0+n] = Vt[d0+n][kt+kk] (contiguous)
    o0 = wmma_bf16(pa, ld_frag_bf16(Vt, 2048, 0,  kt), o0);
    o1 = wmma_bf16(pa, ld_frag_bf16(Vt, 2048, 16, kt), o1);
    o2 = wmma_bf16(pa, ld_frag_bf16(Vt, 2048, 32, kt), o2);
    o3 = wmma_bf16(pa, ld_frag_bf16(Vt, 2048, 48, kt), o3);
  }

  // normalize and store to [b][n][h*64+d] (bf16, feeds projection GEMM)
#pragma unroll
  for (int r = 0; r < 8; ++r) {
    float inv = 1.0f / l[r];
    int npos = row0 + r + grp;
    size_t rowoff = ((size_t)b * 2048 + npos) * 1024 + h * 64;
    ob[rowoff + 0  + col] = (__bf16)(o0[r] * inv);
    ob[rowoff + 16 + col] = (__bf16)(o1[r] * inv);
    ob[rowoff + 32 + col] = (__bf16)(o2[r] * inv);
    ob[rowoff + 48 + col] = (__bf16)(o3[r] * inv);
  }
}

// ---------------- Kernel 3: output projection + bias ---------------------------
// 16x64 tile per wave, A-frag reused across 4 B-frags.
__global__ __launch_bounds__(128) void proj_kernel(
    const __bf16* __restrict__ attn, const __bf16* __restrict__ wpb,
    const float* __restrict__ b_proj, float* __restrict__ out) {
  const int lane = threadIdx.x & 31;
  const int wave = threadIdx.x >> 5;
  const int tile = blockIdx.x * WAVES + wave;        // 0..4095
  const int NT = 16;                                 // 1024/64
  const int mt = tile / NT, nt = tile % NT;
  const int row0 = mt * 16, col0 = nt * 64;

  v8f acc[4] = {};
  for (int k0 = 0; k0 < 1024; k0 += 32) {
    v16bf a = ld_frag_bf16(attn, 1024, row0, k0);
#pragma unroll
    for (int j = 0; j < 4; ++j) {
      v16bf b = ld_frag_bf16(wpb, 1024, col0 + j * 16, k0); // B[k][n]=w_proj[col][k]
      acc[j] = wmma_bf16(a, b, acc[j]);
    }
  }
  const int col = lane & 15;
  const int grp = (lane >> 4) << 3;
#pragma unroll
  for (int j = 0; j < 4; ++j) {
    const int c = col0 + j * 16 + col;
    const float bias = b_proj[c];
#pragma unroll
    for (int r = 0; r < 8; ++r) {
      int row = row0 + r + grp;
      out[(size_t)row * 1024 + c] = acc[j][r] + bias;
    }
  }
}

// ---------------- Host launcher ------------------------------------------------
extern "C" void kernel_launch(void* const* d_in, const int* in_sizes, int n_in,
                              void* d_out, int out_size, void* d_ws, size_t ws_size,
                              hipStream_t stream) {
  (void)in_sizes; (void)n_in; (void)out_size; (void)ws_size;
  const float* x      = (const float*)d_in[0];
  const float* freqs  = (const float*)d_in[1];
  const float* w_qkv  = (const float*)d_in[2];
  const float* w_proj = (const float*)d_in[3];
  const float* b_proj = (const float*)d_in[4];
  float* out = (float*)d_out;

  char* ws = (char*)d_ws;
  const size_t MiB = 1024 * 1024;
  __bf16* qb   = (__bf16*)(ws);                      //  8 MiB  q  [b][h][n][d]
  __bf16* kb   = (__bf16*)(ws +  8 * MiB);           //  8 MiB  k  [b][h][n][d]
  __bf16* vbT  = (__bf16*)(ws + 16 * MiB);           //  8 MiB  v  [b][h][d][n]
  __bf16* ab   = (__bf16*)(ws + 24 * MiB);           //  8 MiB  attn out [b][n][c]
  __bf16* xb   = (__bf16*)(ws + 32 * MiB);           //  8 MiB  x   bf16
  __bf16* wqb  = (__bf16*)(ws + 40 * MiB);           //  6 MiB  w_qkv bf16
  __bf16* wpb  = (__bf16*)(ws + 46 * MiB);           //  2 MiB  w_proj bf16

  // one-shot conversions: x 4Mi, w_qkv 3Mi, w_proj 1Mi elements (8 el/thread)
  hipLaunchKernelGGL(cvt_bf16_kernel, dim3(2048), dim3(256), 0, stream,
                     x, xb, 4194304 / 8);
  hipLaunchKernelGGL(cvt_bf16_kernel, dim3(1536), dim3(256), 0, stream,
                     w_qkv, wqb, 3145728 / 8);
  hipLaunchKernelGGL(cvt_bf16_kernel, dim3(512), dim3(256), 0, stream,
                     w_proj, wpb, 1048576 / 8);

  dim3 blk(128);                                     // 4 waves / WG (wave32)
  hipLaunchKernelGGL(qkv_rope_kernel,   dim3(3072), blk, 0, stream,
                     xb, freqs, wqb, qb, kb, vbT);
  hipLaunchKernelGGL(flash_attn_kernel, dim3(1024), blk, 0, stream,
                     qb, kb, vbT, ab);
  hipLaunchKernelGGL(proj_kernel,       dim3(1024), blk, 0, stream,
                     ab, wpb, b_proj, out);
}
